// LocalPoolPointnet_3813930959054
// MI455X (gfx1250) — matrix operand order repaired and checked by reading, not compiled, not run on hardware
//
#include <hip/hip_runtime.h>
#include <hip/hip_bf16.h>
#include <math.h>

typedef __bf16 bf16_t;
typedef __attribute__((ext_vector_type(16))) __bf16 v16bf;
typedef __attribute__((ext_vector_type(8)))  __bf16 v8bf;
typedef __attribute__((ext_vector_type(8)))  float  v8f;

// CDNA5 hardware tanh (TRANS op). ISA §7.4.3: one independent op / V_NOP
// required after a TRANS op before its output is consumed — bake it in since
// the compiler treats inline asm as a black box.
__device__ __forceinline__ float tanh_hw(float x) {
    float y;
    asm("v_tanh_f32 %0, %1\n\tv_nop" : "=v"(y) : "v"(x));
    return y;
}

__device__ __forceinline__ float gelu_f(float x) {
    // tanh-approximate GELU (matches jax.nn.gelu approximate=True)
    float x3 = x * x * x;
    return 0.5f * x * (1.0f + tanh_hw(0.7978845608028654f * (x + 0.044715f * x3)));
}

// ---------------------------------------------------------------------------
// Repack a K x N f32 weight matrix into WMMA B-fragment order (bf16).
// Fragment f = kt*NT + nt ; element [f*512 + lane*16 + j] = W[kt*32 + k0 + (j<8 ? j : j+8)][nt*16 + lane%16]
// with k0 = (lane<16 ? 0 : 8).  (16-bit B layout, CDNA5 ISA §7.12.2)
// ---------------------------------------------------------------------------
__global__ void repack_b_kernel(const float* __restrict__ W, bf16_t* __restrict__ out,
                                int K, int N) {
    int t = blockIdx.x * blockDim.x + threadIdx.x;
    int NT = N >> 4;
    int KT = K >> 5;
    int total = KT * NT * 32;
    if (t >= total) return;
    int lane = t & 31;
    int f    = t >> 5;
    int nt   = f % NT;
    int kt   = f / NT;
    int col  = nt * 16 + (lane & 15);
    int krow = kt * 32 + ((lane >> 4) ? 8 : 0);
    bf16_t* dst = out + (size_t)f * 512 + lane * 16;
#pragma unroll
    for (int j = 0; j < 16; ++j)
        dst[j] = (bf16_t)W[(size_t)(krow + (j < 8 ? j : j + 8)) * N + col];
}

// ---------------------------------------------------------------------------
// Prep: per point -> pt (trilinear offset), flat voxel index (binary search of
// lin code in the per-batch sorted sparse list), and X0 = pt @ W_pos + b_pos.
// ---------------------------------------------------------------------------
__global__ void prep_kernel(const float* __restrict__ p, const int* __restrict__ sc,
                            const float* __restrict__ Wpos, const float* __restrict__ bpos,
                            const int* __restrict__ resp,
                            int* __restrict__ flat, bf16_t* __restrict__ X0,
                            int npts, int per, int Rr, int npb) {
    int i = blockIdx.x * blockDim.x + threadIdx.x;
    if (i >= npts) return;
    int b = i / npb;
    int res = resp[0];
    float resf = (float)res;
    float pt[3];
    int   cl[3];
#pragma unroll
    for (int d = 0; d < 3; ++d) {
        float c = fminf(fmaxf(p[3 * (size_t)i + d] + 0.5f, 1e-6f), 1.0f - 1e-6f) * resf;
        int ci = (int)c;                 // positive -> trunc == floor
        cl[d] = ci;
        pt[d] = 2.0f * (c - (floorf(c) + 0.5f));
    }
    int lin = (cl[0] * res + cl[1]) * res + cl[2];
    // lower_bound in sorted slin of this batch
    int lo = 0, hi = per;
    const int* base = sc + 4 * (size_t)(b * per);
    while (lo < hi) {
        int mid = (lo + hi) >> 1;
        const int* r = base + 4 * (size_t)mid;
        int sval = (r[1] * res + r[2]) * res + r[3];
        if (sval < lin) lo = mid + 1; else hi = mid;
    }
    flat[i] = b * Rr + lo;
    float p0 = pt[0], p1 = pt[1], p2 = pt[2];
    bf16_t* xrow = X0 + (size_t)i * 256;
#pragma unroll 4
    for (int c = 0; c < 256; ++c) {
        float acc = bpos[c] + p0 * Wpos[c] + p1 * Wpos[256 + c] + p2 * Wpos[512 + c];
        xrow[c] = (bf16_t)acc;
    }
}

// ---------------------------------------------------------------------------
// WMMA helpers: build 16-bit A fragment for row M, k-tile base kb (already
// includes half*8 offset).  v[j] = A[M][kb + j], v[8+j] = A[M][kb + 16 + j].
// ---------------------------------------------------------------------------
__device__ __forceinline__ v16bf make_a_raw(const bf16_t* row, int kb) {
    v8bf lo = *(const v8bf*)(row + kb);
    v8bf hi = *(const v8bf*)(row + kb + 16);
    v16bf a;
#pragma unroll
    for (int j = 0; j < 8; ++j) { a[j] = lo[j]; a[8 + j] = hi[j]; }
    return a;
}
__device__ __forceinline__ v16bf make_a_gelu(const bf16_t* row, int kb) {
    v8bf lo = *(const v8bf*)(row + kb);
    v8bf hi = *(const v8bf*)(row + kb + 16);
    v16bf a;
#pragma unroll
    for (int j = 0; j < 8; ++j) {
        a[j]     = (bf16_t)gelu_f((float)lo[j]);
        a[8 + j] = (bf16_t)gelu_f((float)hi[j]);
    }
    return a;
}

// ---------------------------------------------------------------------------
// Resblock: Y[:,0:128] = X @ Ws + ( gelu( gelu(X)@W0 + b0 ) @ W1 + b1 )
// X: npts x 256 bf16 ; Y written into a width-256 buffer, cols 0..127.
// 256 threads = 8 waves; each wave = one 16-row tile, all 128 output cols.
// ---------------------------------------------------------------------------
__global__ __launch_bounds__(256) void resblock_kernel(
    const bf16_t* __restrict__ X, bf16_t* __restrict__ Y,
    const bf16_t* __restrict__ W0p, const float* __restrict__ b0,
    const bf16_t* __restrict__ W1p, const float* __restrict__ b1,
    const bf16_t* __restrict__ Wsp, int nrows) {
    __shared__ bf16_t hbuf[128 * 128];   // gelu(h), 32 KB
    const int wave = threadIdx.x >> 5;
    const int lane = threadIdx.x & 31;
    const int nsub = lane & 15;
    const int half = lane >> 4;
    const int mo   = half * 8;
    const int rowbase = (blockIdx.x * 8 + wave) * 16;
    const bool active = rowbase < nrows;
    v8f acc[8];

    if (active) {
        const int M = rowbase + nsub;
        const bf16_t* xrow = X + (size_t)M * 256;
        // -------- phase 1: H = gelu(X) @ W0 + b0  (K = 256) --------
#pragma unroll
        for (int nt = 0; nt < 8; ++nt) {
            float bv = b0[nt * 16 + nsub];
#pragma unroll
            for (int r = 0; r < 8; ++r) acc[nt][r] = bv;
        }
#pragma unroll
        for (int kt = 0; kt < 8; ++kt) {
            v16bf a = make_a_gelu(xrow, kt * 32 + half * 8);
#pragma unroll
            for (int nt = 0; nt < 8; ++nt) {
                v16bf bm = *(const v16bf*)(W0p + ((size_t)(kt * 8 + nt) * 32 + lane) * 16);
                acc[nt] = __builtin_amdgcn_wmma_f32_16x16x32_bf16(
                    false, a, false, bm, (short)0, acc[nt], false, false);
            }
        }
        // gelu(H) -> LDS (D layout -> row-major bf16 tile)
#pragma unroll
        for (int nt = 0; nt < 8; ++nt)
#pragma unroll
            for (int r = 0; r < 8; ++r)
                hbuf[(wave * 16 + mo + r) * 128 + nt * 16 + nsub] =
                    (bf16_t)gelu_f(acc[nt][r]);
    }
    __syncthreads();
    if (active) {
        const int M = rowbase + nsub;
        const bf16_t* hrow = hbuf + (size_t)(wave * 16 + nsub) * 128;
        // -------- phase 2: DX = gelu(H) @ W1 + b1  (K = 128) --------
#pragma unroll
        for (int nt = 0; nt < 8; ++nt) {
            float bv = b1[nt * 16 + nsub];
#pragma unroll
            for (int r = 0; r < 8; ++r) acc[nt][r] = bv;
        }
#pragma unroll
        for (int kt = 0; kt < 4; ++kt) {
            v16bf a = make_a_raw(hrow, kt * 32 + half * 8);
#pragma unroll
            for (int nt = 0; nt < 8; ++nt) {
                v16bf bm = *(const v16bf*)(W1p + ((size_t)(kt * 8 + nt) * 32 + lane) * 16);
                acc[nt] = __builtin_amdgcn_wmma_f32_16x16x32_bf16(
                    false, a, false, bm, (short)0, acc[nt], false, false);
            }
        }
        // -------- phase 3: += X @ Ws  (K = 256, skip path) --------
        const bf16_t* xrow = X + (size_t)M * 256;
#pragma unroll
        for (int kt = 0; kt < 8; ++kt) {
            v16bf a = make_a_raw(xrow, kt * 32 + half * 8);
#pragma unroll
            for (int nt = 0; nt < 8; ++nt) {
                v16bf bm = *(const v16bf*)(Wsp + ((size_t)(kt * 8 + nt) * 32 + lane) * 16);
                acc[nt] = __builtin_amdgcn_wmma_f32_16x16x32_bf16(
                    false, a, false, bm, (short)0, acc[nt], false, false);
            }
        }
        // store D (bf16) into cols 0..127 of the width-256 output buffer
#pragma unroll
        for (int nt = 0; nt < 8; ++nt)
#pragma unroll
            for (int r = 0; r < 8; ++r)
                Y[(size_t)(rowbase + mo + r) * 256 + nt * 16 + nsub] = (bf16_t)acc[nt][r];
    }
}

// ---------------------------------------------------------------------------
// Final projection: Y = X[:,0:128] @ Wc + bc   (K = 128, out width 128, no gelu)
// ---------------------------------------------------------------------------
__global__ __launch_bounds__(256) void gemm128_kernel(
    const bf16_t* __restrict__ X, bf16_t* __restrict__ Y,
    const bf16_t* __restrict__ Wp, const float* __restrict__ bias, int nrows) {
    const int wave = threadIdx.x >> 5;
    const int lane = threadIdx.x & 31;
    const int nsub = lane & 15;
    const int half = lane >> 4;
    const int mo   = half * 8;
    const int rowbase = (blockIdx.x * 8 + wave) * 16;
    if (rowbase >= nrows) return;
    const int M = rowbase + nsub;
    const bf16_t* xrow = X + (size_t)M * 256;   // net lives in cols 0..127
    v8f acc[8];
#pragma unroll
    for (int nt = 0; nt < 8; ++nt) {
        float bv = bias[nt * 16 + nsub];
#pragma unroll
        for (int r = 0; r < 8; ++r) acc[nt][r] = bv;
    }
#pragma unroll
    for (int kt = 0; kt < 4; ++kt) {
        v16bf a = make_a_raw(xrow, kt * 32 + half * 8);
#pragma unroll
        for (int nt = 0; nt < 8; ++nt) {
            v16bf bm = *(const v16bf*)(Wp + ((size_t)(kt * 8 + nt) * 32 + lane) * 16);
            acc[nt] = __builtin_amdgcn_wmma_f32_16x16x32_bf16(
                false, a, false, bm, (short)0, acc[nt], false, false);
        }
    }
#pragma unroll
    for (int nt = 0; nt < 8; ++nt)
#pragma unroll
        for (int r = 0; r < 8; ++r)
            Y[(size_t)(rowbase + mo + r) * 128 + nt * 16 + nsub] = (bf16_t)acc[nt][r];
}

// ---------------------------------------------------------------------------
// Pooling
// ---------------------------------------------------------------------------
__global__ void zero_kernel(float* __restrict__ sums, float* __restrict__ cnt,
                            int ns, int nc) {
    int t = blockIdx.x * blockDim.x + threadIdx.x;
    if (t < ns) sums[t] = 0.0f;
    if (t < nc) cnt[t] = 0.0f;
}

__global__ void scatter_kernel(const bf16_t* __restrict__ src, int stride,
                               const int* __restrict__ flat,
                               float* __restrict__ sums, float* __restrict__ cnt,
                               int npts) {
    int t = blockIdx.x * blockDim.x + threadIdx.x;
    if (t >= npts * 128) return;
    int i  = t >> 7;
    int ch = t & 127;
    int f  = flat[i];
    atomicAdd(&sums[(size_t)f * 128 + ch], (float)src[(size_t)i * stride + ch]);
    if (ch == 0) atomicAdd(&cnt[f], 1.0f);
}

__global__ void gather_kernel(bf16_t* __restrict__ dst,   // width-256 buffer, cols 128..255
                              const int* __restrict__ flat,
                              const float* __restrict__ sums, const float* __restrict__ cnt,
                              int npts) {
    int t = blockIdx.x * blockDim.x + threadIdx.x;
    if (t >= npts * 128) return;
    int i  = t >> 7;
    int ch = t & 127;
    int f  = flat[i];
    float v = sums[(size_t)f * 128 + ch] / fmaxf(cnt[f], 1.0f);
    dst[(size_t)i * 256 + 128 + ch] = (bf16_t)v;
}

__global__ void out_kernel(float* __restrict__ out, const int* __restrict__ sc,
                           const float* __restrict__ sums, const float* __restrict__ cnt,
                           int M, int per, int Rr) {
    int t = blockIdx.x * blockDim.x + threadIdx.x;
    if (t >= M * 128) return;
    int j  = t >> 7;
    int ch = t & 127;
    int b  = sc[4 * (size_t)j];
    int pos = j - b * per;
    int f = b * Rr + pos;
    out[t] = sums[(size_t)f * 128 + ch] / fmaxf(cnt[f], 1.0f);
}

// ---------------------------------------------------------------------------
extern "C" void kernel_launch(void* const* d_in, const int* in_sizes, int n_in,
                              void* d_out, int out_size, void* d_ws, size_t ws_size,
                              hipStream_t stream) {
    const float* p    = (const float*)d_in[0];
    const int*   sc   = (const int*)d_in[1];
    const float* Wpos = (const float*)d_in[2];
    const float* bpos = (const float*)d_in[3];
    const float* W0   = (const float*)d_in[4];   // (5,256,128)
    const float* b0   = (const float*)d_in[5];   // (5,128)
    const float* W1   = (const float*)d_in[6];   // (5,128,128)
    const float* b1   = (const float*)d_in[7];
    const float* Ws   = (const float*)d_in[8];   // (5,256,128)
    const float* Wc   = (const float*)d_in[9];   // (128,128)
    const float* bc   = (const float*)d_in[10];
    const int*   resp = (const int*)d_in[11];

    const int npts = in_sizes[0] / 3;    // 200000
    const int Msp  = in_sizes[1] / 4;    // 40000
    const int B    = 2;
    const int per  = Msp / B;            // 20000
    const int Rr   = per + 5;            // 20005
    const int npb  = npts / B;

    // ---- workspace carve (256-byte aligned) ----
    char* w = (char*)d_ws;
    auto carve = [&](size_t bytes) {
        void* r = (void*)w;
        w += (bytes + 255) & ~(size_t)255;
        return r;
    };
    int*    flat = (int*)carve((size_t)npts * 4);
    bf16_t* bufA = (bf16_t*)carve((size_t)npts * 256 * 2);
    bf16_t* bufB = (bf16_t*)carve((size_t)npts * 256 * 2);
    bf16_t* cbuf = (bf16_t*)carve((size_t)npts * 128 * 2);
    float*  sums = (float*)carve((size_t)B * Rr * 128 * 4);
    float*  cnt  = (float*)carve((size_t)B * Rr * 4);
    bf16_t* W0p  = (bf16_t*)carve((size_t)5 * 32768 * 2);  // 8x8 frags * 512
    bf16_t* W1p  = (bf16_t*)carve((size_t)5 * 16384 * 2);  // 4x8 frags * 512
    bf16_t* Wsp  = (bf16_t*)carve((size_t)5 * 32768 * 2);
    bf16_t* Wcp  = (bf16_t*)carve((size_t)16384 * 2);

    // ---- repack weights into B-fragment order ----
    for (int blk = 0; blk < 5; ++blk) {
        repack_b_kernel<<<(2048 + 255) / 256, 256, 0, stream>>>(
            W0 + (size_t)blk * 256 * 128, W0p + (size_t)blk * 32768, 256, 128);
        repack_b_kernel<<<(1024 + 255) / 256, 256, 0, stream>>>(
            W1 + (size_t)blk * 128 * 128, W1p + (size_t)blk * 16384, 128, 128);
        repack_b_kernel<<<(2048 + 255) / 256, 256, 0, stream>>>(
            Ws + (size_t)blk * 256 * 128, Wsp + (size_t)blk * 32768, 256, 128);
    }
    repack_b_kernel<<<(1024 + 255) / 256, 256, 0, stream>>>(Wc, Wcp, 128, 128);

    // ---- prep: pt, flat index, X0 = pt @ W_pos + b_pos ----
    prep_kernel<<<(npts + 255) / 256, 256, 0, stream>>>(
        p, sc, Wpos, bpos, resp, flat, bufA, npts, per, Rr, npb);

    const int tiles = (npts + 15) / 16;
    const int rbBlocks = (tiles + 7) / 8;
    const int nsz = B * Rr * 128;
    const int ncz = B * Rr;
    const int ptBlocks = ((size_t)npts * 128 + 255) / 256;

    // ---- block 0 ----
    resblock_kernel<<<rbBlocks, 256, 0, stream>>>(
        bufA, bufB, W0p, b0, W1p, b1, Wsp, npts);

    // ---- blocks 1..4 with pooling ----
    bf16_t* cur = bufB;
    bf16_t* oth = bufA;
    for (int i = 1; i < 5; ++i) {
        zero_kernel<<<(nsz + 255) / 256, 256, 0, stream>>>(sums, cnt, nsz, ncz);
        scatter_kernel<<<ptBlocks, 256, 0, stream>>>(cur, 256, flat, sums, cnt, npts);
        gather_kernel<<<ptBlocks, 256, 0, stream>>>(cur, flat, sums, cnt, npts);
        resblock_kernel<<<rbBlocks, 256, 0, stream>>>(
            cur, oth,
            W0p + (size_t)i * 32768, b0 + (size_t)i * 128,
            W1p + (size_t)i * 16384, b1 + (size_t)i * 128,
            Wsp + (size_t)i * 32768, npts);
        bf16_t* t = cur; cur = oth; oth = t;
    }
    // cur now holds net4 in cols 0..127

    // ---- c = net @ Wc + b_c ----
    gemm128_kernel<<<rbBlocks, 256, 0, stream>>>(cur, cbuf, Wcp, bc, npts);

    // ---- final scatter-mean of c and output gather ----
    zero_kernel<<<(nsz + 255) / 256, 256, 0, stream>>>(sums, cnt, nsz, ncz);
    scatter_kernel<<<ptBlocks, 256, 0, stream>>>(cbuf, 128, flat, sums, cnt, npts);
    out_kernel<<<((size_t)Msp * 128 + 255) / 256, 256, 0, stream>>>(
        (float*)d_out, sc, sums, cnt, Msp, per, Rr);
}
